// MaskedHead_39719857553632
// MI455X (gfx1250) — compile-verified, matrix operand order checked by reading, not run on hardware
//
#include <hip/hip_runtime.h>
#include <hip/hip_bf16.h>

typedef _Float16 h16;
typedef __attribute__((ext_vector_type(16))) _Float16 v16h;
typedef __attribute__((ext_vector_type(8)))  _Float16 h8;
typedef __attribute__((ext_vector_type(8)))  float    v8f;

// Build a 16x32-f16 A/B fragment for v_wmma_f32_16x16x32_f16 from two
// contiguous 8-halfword (16B) chunks (K-offsets kb and kb+16 per ISA 7.12.2).
__device__ __forceinline__ v16h load_frag(const h16* lo, const h16* hi) {
    h8 a = *(const h8*)lo;
    h8 b = *(const h8*)hi;
    v16h r;
#pragma unroll
    for (int i = 0; i < 8; ++i) { r[i] = a[i]; r[i + 8] = b[i]; }
    return r;
}

__device__ __forceinline__ v8f wmma16(v16h a, v16h b, v8f c) {
    // (neg_a, A, neg_b, B, c_mod, C, reuse_a, reuse_b)
    return __builtin_amdgcn_wmma_f32_16x16x32_f16(false, a, false, b, (short)0, c,
                                                  false, false);
}

#define TT 2048
#define DD 1024
#define HS 64

// ---------------------------------------------------------------- weights
// Wt[n*DD + k] = (f16) W[k*HS + n]  (transpose + downconvert, 64x1024)
__global__ void wtrans_kernel(const float* __restrict__ W, h16* __restrict__ Wt) {
    int idx = blockIdx.x * blockDim.x + threadIdx.x;     // 0 .. 64*1024-1
    if (idx < HS * DD) {
        int n = idx >> 10, k = idx & (DD - 1);
        Wt[idx] = (h16)W[k * HS + n];
    }
}

// ---------------------------------------------------------------- QKV proj
// grid.x = B*T/16 token tiles; 4 waves, wave w owns HS columns [16w,16w+16)
__global__ __launch_bounds__(128) void proj_kernel(
    const float* __restrict__ x,
    const h16* __restrict__ Wtq, const h16* __restrict__ Wtk,
    const h16* __restrict__ Wtv,
    const float* __restrict__ bq, const float* __restrict__ bk,
    h16* __restrict__ qh, h16* __restrict__ kh, h16* __restrict__ vt) {
    __shared__ h16 xs[16 * DD];                          // 32 KB
    const int tile = blockIdx.x;
    const int row0 = tile * 16;                          // global token base
    const int tid  = threadIdx.x;

    for (int i = tid; i < 16 * DD; i += 128) {
        int r = i >> 10, c = i & (DD - 1);
        xs[i] = (h16)x[(size_t)(row0 + r) * DD + c];
    }
    __syncthreads();

    const int wave  = tid >> 5;
    const int lane  = tid & 31;
    const int lhalf = lane >> 4;
    const int l16   = lane & 15;
    const int n     = wave * 16 + l16;                   // HS column

    v8f cq = {}, ck = {}, cv = {};
    const int kofs = lhalf * 8;
    for (int k0 = 0; k0 < DD; k0 += 32) {
        const int kb = k0 + kofs;
        v16h a  = load_frag(&xs[l16 * DD + kb],  &xs[l16 * DD + kb + 16]);
        v16h bfq = load_frag(&Wtq[(size_t)n * DD + kb], &Wtq[(size_t)n * DD + kb + 16]);
        v16h bfk = load_frag(&Wtk[(size_t)n * DD + kb], &Wtk[(size_t)n * DD + kb + 16]);
        v16h bfv = load_frag(&Wtv[(size_t)n * DD + kb], &Wtv[(size_t)n * DD + kb + 16]);
        cq = wmma16(a, bfq, cq);
        ck = wmma16(a, bfk, ck);
        cv = wmma16(a, bfv, cv);
    }
    const float biasq = bq[n], biask = bk[n];
    const int b = row0 / TT;                             // batch index
#pragma unroll
    for (int r = 0; r < 8; ++r) {
        int M   = r + lhalf * 8;                         // C/D row layout
        int tok = row0 + M;                              // global token
        qh[(size_t)tok * HS + n] = (h16)(cq[r] + biasq);
        kh[(size_t)tok * HS + n] = (h16)(ck[r] + biask);
        int t = tok - b * TT;
        vt[((size_t)b * HS + n) * TT + t] = (h16)cv[r];  // V transposed
    }
}

// ---------------------------------------------------------------- attention
// grid.x * 4 waves = B*T/16 query tiles; flash loop, 32 keys per step.
__global__ __launch_bounds__(128) void attn_kernel(
    const h16* __restrict__ qh, const h16* __restrict__ kh,
    const h16* __restrict__ vt, float* __restrict__ out) {
    __shared__ h16 plds[4][16 * 32];                     // 1 KB per wave

    const int wave  = threadIdx.x >> 5;
    const int lane  = threadIdx.x & 31;
    const int lhalf = lane >> 4;
    const int l16   = lane & 15;
    const int kofs  = lhalf * 8;

    const int qtile = blockIdx.x * 4 + wave;             // 0..511
    const int b     = qtile >> 7;                        // 128 tiles / batch
    const int i0    = (qtile & 127) * 16;                // query base in batch
    const size_t tokbase = (size_t)b * TT;

    // Q fragments for hs 0..31 and 32..63 (A layout: row = l16)
    const h16* qrow = &qh[(tokbase + i0 + l16) * HS];
    v16h qa0 = load_frag(qrow + kofs,      qrow + kofs + 16);
    v16h qa1 = load_frag(qrow + 32 + kofs, qrow + 32 + kofs + 16);

    v8f o0 = {}, o1 = {}, o2 = {}, o3 = {};
    float m[8], l[8];
#pragma unroll
    for (int r = 0; r < 8; ++r) { m[r] = -3.0e38f; l[r] = 0.0f; }
    const float scale = 8.0f;                            // sqrt(HS): faithful to ref bug
    h16* pw = &plds[wave][0];

    for (int j0 = 0; j0 <= i0 + 15; j0 += 32) {
        // ---- S = Q K^T for 2 subtiles of 16 keys
        v8f s[2];
#pragma unroll
        for (int sub = 0; sub < 2; ++sub) {
            const h16* krow = &kh[(tokbase + j0 + sub * 16 + l16) * HS];
            v16h kb0 = load_frag(krow + kofs,      krow + kofs + 16);
            v16h kb1 = load_frag(krow + 32 + kofs, krow + 32 + kofs + 16);
            v8f c = {};
            c = wmma16(qa0, kb0, c);
            c = wmma16(qa1, kb1, c);
            s[sub] = c;
        }
        // ---- scale + causal mask + row max (across each 16-lane half)
        float alpha[8];
#pragma unroll
        for (int r = 0; r < 8; ++r) {
            const int qg = i0 + r + lhalf * 8;           // query idx in batch
            float v0 = s[0][r] * scale;
            float v1 = s[1][r] * scale;
            if (j0 + l16 > qg)      v0 = -3.0e38f;
            if (j0 + 16 + l16 > qg) v1 = -3.0e38f;
            s[0][r] = v0; s[1][r] = v1;
            float mx = fmaxf(v0, v1);
#pragma unroll
            for (int off = 1; off < 16; off <<= 1)
                mx = fmaxf(mx, __shfl_xor(mx, off, 32));
            float mnew = fmaxf(m[r], mx);
            alpha[r] = __expf(m[r] - mnew);
            m[r] = mnew;
        }
        // ---- P = exp(S - m), row sums, stash P in LDS (D-layout -> A-layout)
#pragma unroll
        for (int r = 0; r < 8; ++r) {
            float p0 = __expf(s[0][r] - m[r]);
            float p1 = __expf(s[1][r] - m[r]);
            float rs = p0 + p1;
#pragma unroll
            for (int off = 1; off < 16; off <<= 1)
                rs += __shfl_xor(rs, off, 32);
            l[r] = l[r] * alpha[r] + rs;
            const int M = r + lhalf * 8;
            pw[M * 32 + l16]      = (h16)p0;
            pw[M * 32 + 16 + l16] = (h16)p1;
        }
        asm volatile("s_wait_dscnt 0" ::: "memory");     // wave-local LDS RAW
        v16h pa = load_frag(&pw[l16 * 32 + kofs], &pw[l16 * 32 + kofs + 16]);

        // ---- rescale O and accumulate P·V (V is [b][hs][t] -> contiguous)
#pragma unroll
        for (int r = 0; r < 8; ++r) {
            o0[r] *= alpha[r]; o1[r] *= alpha[r];
            o2[r] *= alpha[r]; o3[r] *= alpha[r];
        }
        const h16* v0r = &vt[((size_t)b * HS + 0  + l16) * TT + j0];
        const h16* v1r = &vt[((size_t)b * HS + 16 + l16) * TT + j0];
        const h16* v2r = &vt[((size_t)b * HS + 32 + l16) * TT + j0];
        const h16* v3r = &vt[((size_t)b * HS + 48 + l16) * TT + j0];
        o0 = wmma16(pa, load_frag(v0r + kofs, v0r + kofs + 16), o0);
        o1 = wmma16(pa, load_frag(v1r + kofs, v1r + kofs + 16), o1);
        o2 = wmma16(pa, load_frag(v2r + kofs, v2r + kofs + 16), o2);
        o3 = wmma16(pa, load_frag(v3r + kofs, v3r + kofs + 16), o3);

        // prefetch next key tile
        __builtin_prefetch(&kh[(tokbase + j0 + 32 + l16) * HS], 0, 1);
    }
    // ---- normalize and store fp32
#pragma unroll
    for (int r = 0; r < 8; ++r) {
        const float inv = 1.0f / l[r];
        const int M = r + lhalf * 8;
        const size_t rowoff = (tokbase + i0 + M) * HS;
        out[rowoff +  0 + l16] = o0[r] * inv;
        out[rowoff + 16 + l16] = o1[r] * inv;
        out[rowoff + 32 + l16] = o2[r] * inv;
        out[rowoff + 48 + l16] = o3[r] * inv;
    }
}

extern "C" void kernel_launch(void* const* d_in, const int* in_sizes, int n_in,
                              void* d_out, int out_size, void* d_ws, size_t ws_size,
                              hipStream_t stream) {
    (void)in_sizes; (void)n_in; (void)out_size; (void)ws_size;
    const float* x  = (const float*)d_in[0];
    const float* Wk = (const float*)d_in[1];
    const float* bk = (const float*)d_in[2];
    const float* Wq = (const float*)d_in[3];
    const float* bq = (const float*)d_in[4];
    const float* Wv = (const float*)d_in[5];

    char* ws = (char*)d_ws;
    h16* qh  = (h16*)(ws + (size_t)0);                   // 1 MB
    h16* kh  = (h16*)(ws + ((size_t)1 << 20));           // 1 MB
    h16* vt  = (h16*)(ws + ((size_t)2 << 20));           // 1 MB (transposed V)
    h16* wtq = (h16*)(ws + ((size_t)3 << 20));           // 128 KB each
    h16* wtk = (h16*)(ws + ((size_t)3 << 20) + (128 << 10));
    h16* wtv = (h16*)(ws + ((size_t)3 << 20) + (256 << 10));

    wtrans_kernel<<<(HS * DD + 255) / 256, 256, 0, stream>>>(Wq, wtq);
    wtrans_kernel<<<(HS * DD + 255) / 256, 256, 0, stream>>>(Wk, wtk);
    wtrans_kernel<<<(HS * DD + 255) / 256, 256, 0, stream>>>(Wv, wtv);

    // 4*2048/16 = 512 token tiles, 4 waves each (one HS column tile per wave)
    proj_kernel<<<512, 128, 0, stream>>>(x, wtq, wtk, wtv, bq, bk, qh, kh, vt);

    // 512 query tiles, 4 waves per block -> 128 blocks
    attn_kernel<<<128, 128, 0, stream>>>(qh, kh, vt, (float*)d_out);
}
